// LSTM_7378753815405
// MI455X (gfx1250) — matrix operand order repaired and checked by reading, not compiled, not run on hardware
//
#include <hip/hip_runtime.h>
#include <hip/hip_bf16.h>

typedef __attribute__((ext_vector_type(16))) _Float16 v16h;
typedef __attribute__((ext_vector_type(8)))  float    v8f;

// Problem sizes
#define NB   16
#define CIN  64
#define COUT 128
#define LSEQ 512
#define TSTEPS 64

// LDS layout (dynamic shared; 320KB per WGP available on CDNA5)
constexpr int BX_OFF   = 0;        // 4g x 8n x 2kk x 1KB = 64 KB  (x-weight B fragments, f16)
constexpr int BH_OFF   = 65536;    // 4g x 8n x 4kk x 1KB = 128 KB (h-weight B fragments, f16)
constexpr int AX_OFF   = 196608;   // 2kk x 1KB = 2 KB             (x activation A fragments)
constexpr int AH_OFF   = 198656;   // 4kk x 1KB = 4 KB             (h activation A fragments)
constexpr int PART_OFF = 202752;   // 4q x 16row x 8wave x 4B = 2 KB (LN partial sums)
constexpr int STAT_OFF = 204800;   // mu[2][16], rstd[2][16] = 256 B
constexpr int SMEM_BYTES = 205056;

__device__ __forceinline__ float fast_sig(float x) {
  float e = __builtin_amdgcn_exp2f(-1.4426950408889634f * x);
  return __builtin_amdgcn_rcpf(1.0f + e);
}
__device__ __forceinline__ float fast_tanh(float x) {
  float e = __builtin_amdgcn_exp2f(2.8853900817779268f * x); // exp(2x)
  return 1.0f - 2.0f * __builtin_amdgcn_rcpf(e + 1.0f);
}

__global__ void __launch_bounds__(256)
lstm_wmma(const float* __restrict__ x,
          const float* __restrict__ Wxi, const float* __restrict__ bxi,
          const float* __restrict__ Whi, const float* __restrict__ bhi,
          const float* __restrict__ Wxf, const float* __restrict__ bxf,
          const float* __restrict__ Whf, const float* __restrict__ bhf,
          const float* __restrict__ Wxc, const float* __restrict__ bxc,
          const float* __restrict__ Whc, const float* __restrict__ bhc,
          const float* __restrict__ Wxo, const float* __restrict__ bxo,
          const float* __restrict__ Who, const float* __restrict__ bho,
          const float* __restrict__ lng, const float* __restrict__ lnb,
          const float* __restrict__ lncg, const float* __restrict__ lncb,
          float* __restrict__ out)
{
  extern __shared__ char smem[];
  const int tid  = threadIdx.x;
  const int lane = tid & 31;
  const int wv   = tid >> 5;          // wave id 0..7, owns output channels wv*16..wv*16+15
  const int wg   = blockIdx.x;        // tile of 16 (b,l) positions

  // ---------------- Pack weights into LDS as WMMA B-fragments (f16) ----------------
  // B-frag layout (16x16x32 f16): lanes 0-15 hold K 0..7 then 16..23, lanes 16-31 hold K 8..15 then 24..31,
  // N = lane&15, matching the A-fragment K order.
#pragma unroll
  for (int g = 0; g < 4; ++g) {
    const float* WXg = (g == 0) ? Wxi : (g == 1) ? Wxf : (g == 2) ? Wxc : Wxo;
    const float* WHg = (g == 0) ? Whi : (g == 1) ? Whf : (g == 2) ? Whc : Who;
    // x-side: 128x64 = 8192 halves per gate
    for (int k = 0; k < 32; ++k) {
      int e = tid + k * 256;
      int frag = e >> 9, lt = (e >> 4) & 31, m = e & 15;
      int n = frag >> 1, kk = frag & 1;
      int o = n * 16 + (lt & 15);
      int K = kk * 32 + ((m < 8) ? ((lt >> 4) * 8 + m) : (16 + (lt >> 4) * 8 + (m - 8)));
      *(_Float16*)(smem + BX_OFF + ((g * 8 + n) * 2 + kk) * 1024 + lt * 32 + m * 2) =
          (_Float16)WXg[o * CIN + K];
    }
    // h-side: 128x128 = 16384 halves per gate
    for (int k = 0; k < 64; ++k) {
      int e = tid + k * 256;
      int frag = e >> 9, lt = (e >> 4) & 31, m = e & 15;
      int n = frag >> 2, kk = frag & 3;
      int o = n * 16 + (lt & 15);
      int K = kk * 32 + ((m < 8) ? ((lt >> 4) * 8 + m) : (16 + (lt >> 4) * 8 + (m - 8)));
      *(_Float16*)(smem + BH_OFF + ((g * 8 + n) * 4 + kk) * 1024 + lt * 32 + m * 2) =
          (_Float16)WHg[o * COUT + K];
    }
  }

  // ---------------- Per-lane persistent state ----------------
  const int ch = wv * 16 + (lane & 15);       // output channel this lane produces
  float bX[4] = { bxi[ch], bxf[ch], bxc[ch], bxo[ch] };
  float bHH[4] = { bhi[ch], bhf[ch], bhc[ch], bho[ch] };
  const float gH = lng[ch],  bHn = lnb[ch];
  const float gC = lncg[ch], bCn = lncb[ch];

  // A-fragment store coordinates for writing h back (per-lane constant channel)
  const int c2  = ch & 31;
  const int kkH = ch >> 5;
  const int mA  = (c2 < 8) ? c2 : (c2 < 16) ? (c2 - 8) : (c2 < 24) ? (8 + c2 - 16) : (8 + c2 - 24);
  const int hiA = ((c2 >= 8 && c2 < 16) || (c2 >= 24)) ? 16 : 0;

  float c_reg[8];
#pragma unroll
  for (int r = 0; r < 8; ++r) c_reg[r] = 0.0f;

  __syncthreads();  // weights packed

  // ---------------- Recurrence over timesteps ----------------
  for (int t = 0; t < TSTEPS; ++t) {
    const bool first = (t == 0);

    // Stage x[:, :, :, t] tile (16 pos x 64 ch) into LDS in A-fragment layout (f16).
    // A layout: lanes 0-15: K 0..7 then 16..23 ; lanes 16-31: K 8..15 then 24..31.
#pragma unroll
    for (int k2 = 0; k2 < 4; ++k2) {
      int e = tid + k2 * 256;                 // 1024 halves total
      int kk = e >> 9, lt = (e >> 4) & 31, m = e & 15;
      int row = lt & 15, hi = lt >> 4;
      int K = kk * 32 + ((m < 8) ? (hi * 8 + m) : (16 + hi * 8 + (m - 8)));
      int p = wg * 16 + row;
      int b = p >> 9, l = p & 511;
      float xv = x[((b * CIN + K) * LSEQ + l) * TSTEPS + t];
      *(_Float16*)(smem + AX_OFF + e * 2) = (_Float16)xv;
    }
    __syncthreads();

    // ---- Gate matmuls: acc[g] = bias + x_t @ Wx_g^T (+ h @ Wh_g^T) ----
    v8f acc[4];
#pragma unroll
    for (int g = 0; g < 4; ++g) {
      float bb = bX[g] + (first ? 0.0f : bHH[g]);
#pragma unroll
      for (int r = 0; r < 8; ++r) acc[g][r] = bb;
    }
#pragma unroll
    for (int kk = 0; kk < 2; ++kk) {
      v16h a = *(const v16h*)(smem + AX_OFF + kk * 1024 + lane * 32);
#pragma unroll
      for (int g = 0; g < 4; ++g) {
        v16h bf = *(const v16h*)(smem + BX_OFF + ((g * 8 + wv) * 2 + kk) * 1024 + lane * 32);
        acc[g] = __builtin_amdgcn_wmma_f32_16x16x32_f16(false, a, false, bf,
                                                        (short)0, acc[g], false, false);
      }
    }
    if (!first) {
#pragma unroll
      for (int kk = 0; kk < 4; ++kk) {
        v16h a = *(const v16h*)(smem + AH_OFF + kk * 1024 + lane * 32);
#pragma unroll
        for (int g = 0; g < 4; ++g) {
          v16h bf = *(const v16h*)(smem + BH_OFF + ((g * 8 + wv) * 4 + kk) * 1024 + lane * 32);
          acc[g] = __builtin_amdgcn_wmma_f32_16x16x32_f16(false, a, false, bf,
                                                          (short)0, acc[g], false, false);
        }
      }
    }

    // ---- Nonlinearity + cell update + LN partial sums ----
    float hraw[8], cnew[8];
#pragma unroll
    for (int r = 0; r < 8; ++r) {
      float ig = fast_sig(acc[0][r]);
      float fg = fast_sig(acc[1][r]);
      float cg = fast_tanh(acc[2][r]);
      float og = fast_sig(acc[3][r]);
      float cn = c_reg[r] * fg + ig * cg;     // at t=0: c_reg=0 -> ig*cg, matching reference
      float hr = fast_tanh(cn) * og;
      cnew[r] = cn; hraw[r] = hr;

      float sh = hr, qh = hr * hr, sc = cn, qc = cn * cn;
#pragma unroll
      for (int msk = 1; msk < 16; msk <<= 1) {
        sh += __shfl_xor(sh, msk, 16);
        qh += __shfl_xor(qh, msk, 16);
        sc += __shfl_xor(sc, msk, 16);
        qc += __shfl_xor(qc, msk, 16);
      }
      if ((lane & 15) == 0) {
        int row = (lane >> 4) * 8 + r;
        *(float*)(smem + PART_OFF + ((0 * 16 + row) * 8 + wv) * 4) = sh;
        *(float*)(smem + PART_OFF + ((1 * 16 + row) * 8 + wv) * 4) = qh;
        *(float*)(smem + PART_OFF + ((2 * 16 + row) * 8 + wv) * 4) = sc;
        *(float*)(smem + PART_OFF + ((3 * 16 + row) * 8 + wv) * 4) = qc;
      }
    }
    __syncthreads();

    // ---- Finalize LN statistics (2 quantities x 16 rows) ----
    if (tid < 32) {
      int which = tid >> 4, row = tid & 15;   // which: 0=h, 1=c
      float s = 0.0f, s2 = 0.0f;
#pragma unroll
      for (int w2 = 0; w2 < 8; ++w2) {
        s  += *(const float*)(smem + PART_OFF + (((which * 2 + 0) * 16 + row) * 8 + w2) * 4);
        s2 += *(const float*)(smem + PART_OFF + (((which * 2 + 1) * 16 + row) * 8 + w2) * 4);
      }
      float mu  = s * 0.0078125f;                      // /128
      float var = s2 * 0.0078125f - mu * mu;
      float rs  = __builtin_amdgcn_rsqf(var + 1e-5f);
      *(float*)(smem + STAT_OFF +       (which * 16 + row) * 4) = mu;
      *(float*)(smem + STAT_OFF + 128 + (which * 16 + row) * 4) = rs;
    }
    __syncthreads();

    // ---- Normalize, emit output, write h back as next A-fragments ----
#pragma unroll
    for (int r = 0; r < 8; ++r) {
      int row = (lane >> 4) * 8 + r;
      float muh = *(const float*)(smem + STAT_OFF + row * 4);
      float rsh = *(const float*)(smem + STAT_OFF + 128 + row * 4);
      float muc = *(const float*)(smem + STAT_OFF + (16 + row) * 4);
      float rsc = *(const float*)(smem + STAT_OFF + 128 + (16 + row) * 4);
      float hn = (hraw[r] - muh) * rsh * gH + bHn;
      float cn = (cnew[r] - muc) * rsc * gC + bCn;
      c_reg[r] = cn;

      int p = wg * 16 + row;
      int b = p >> 9, l = p & 511;
      out[(b * COUT + ch) * (LSEQ * TSTEPS) + l * TSTEPS + t] = hn;
      *(_Float16*)(smem + AH_OFF + kkH * 1024 + (hiA + row) * 32 + mA * 2) = (_Float16)hn;
    }
    // next iteration's post-staging barrier makes AH writes visible before reads
  }
}

extern "C" void kernel_launch(void* const* d_in, const int* in_sizes, int n_in,
                              void* d_out, int out_size, void* d_ws, size_t ws_size,
                              hipStream_t stream) {
  const float* x    = (const float*)d_in[0];
  const float* Wii  = (const float*)d_in[1];
  const float* bii  = (const float*)d_in[2];
  const float* Wih  = (const float*)d_in[3];
  const float* bih  = (const float*)d_in[4];
  const float* Wfi  = (const float*)d_in[5];
  const float* bfi  = (const float*)d_in[6];
  const float* Wfh  = (const float*)d_in[7];
  const float* bfh  = (const float*)d_in[8];
  const float* Wci  = (const float*)d_in[9];
  const float* bci  = (const float*)d_in[10];
  const float* Wch  = (const float*)d_in[11];
  const float* bch  = (const float*)d_in[12];
  const float* Woi  = (const float*)d_in[13];
  const float* boi  = (const float*)d_in[14];
  const float* Woh  = (const float*)d_in[15];
  const float* boh  = (const float*)d_in[16];
  const float* lng  = (const float*)d_in[17];
  const float* lnb  = (const float*)d_in[18];
  const float* lncg = (const float*)d_in[19];
  const float* lncb = (const float*)d_in[20];

  (void)hipFuncSetAttribute((const void*)lstm_wmma,
                            hipFuncAttributeMaxDynamicSharedMemorySize, SMEM_BYTES);

  // 8192 positions / 16 per workgroup = 512 workgroups, 8 waves each
  lstm_wmma<<<512, 256, SMEM_BYTES, stream>>>(
      x, Wii, bii, Wih, bih, Wfi, bfi, Wfh, bfh,
      Wci, bci, Wch, bch, Woi, boi, Woh, boh,
      lng, lnb, lncg, lncb, (float*)d_out);
}
